// WaveNet_27650999452587
// MI455X (gfx1250) — matrix operand order repaired and checked by reading, not compiled, not run on hardware
//
#include <hip/hip_runtime.h>

// WaveNet on MI455X (gfx1250): bf16 WMMA (v_wmma_f32_16x16x32_bf16) for the
// dilated + residual convs, bf16 [B][T][C] time-major activations (B-operand
// friendly), ping-pong buffers across the 20 sequential layer launches,
// post-conv folded into a running skip accumulator in d_out.
// Round 2: 32 timesteps per wave (A-operand reuse, 28 WMMAs/wave) and
// hardware-trans gating (v_exp_f32 + v_rcp_f32 instead of ocml tanhf/expf).

#define T_LEN   65536
#define NCH     32
#define NGATED  64
#define NLAYERS 20

typedef __attribute__((ext_vector_type(16))) __bf16       v16bf;
typedef __attribute__((ext_vector_type(8)))  float        v8f;
typedef __attribute__((ext_vector_type(8)))  unsigned int v8u;

__device__ __forceinline__ unsigned int pack2_bf16(float a, float b) {
    unsigned int ua = __builtin_bit_cast(unsigned int, a);
    unsigned int ub = __builtin_bit_cast(unsigned int, b);
    ua += 0x7FFFu + ((ua >> 16) & 1u);   // round-to-nearest-even
    ub += 0x7FFFu + ((ub >> 16) & 1u);
    return (ua >> 16) | (ub & 0xFFFF0000u);
}

__device__ __forceinline__ unsigned short cvt1_bf16(float a) {
    unsigned int ua = __builtin_bit_cast(unsigned int, a);
    ua += 0x7FFFu + ((ua >> 16) & 1u);
    return (unsigned short)(ua >> 16);
}

__device__ __forceinline__ float bf16lo(unsigned int w) {
    return __builtin_bit_cast(float, w << 16);
}
__device__ __forceinline__ float bf16hi(unsigned int w) {
    return __builtin_bit_cast(float, w & 0xFFFF0000u);
}

#define LOG2E 1.44269504088896340736f

// sigmoid(x) = rcp(1 + exp2(-x*log2e)) : one v_exp_f32 + one v_rcp_f32
__device__ __forceinline__ float hw_sigmoid(float x) {
    float e = __builtin_amdgcn_exp2f(x * -LOG2E);
    return __builtin_amdgcn_rcpf(1.0f + e);
}
// tanh(x) = 2*sigmoid(2x) - 1
__device__ __forceinline__ float hw_tanh(float x) {
    float e = __builtin_amdgcn_exp2f(x * (-2.0f * LOG2E));
    return __builtin_fmaf(2.0f, __builtin_amdgcn_rcpf(1.0f + e), -1.0f);
}

// ---------------------------------------------------------------------------
// Cast weights to bf16. Wd[l][g][c + 32*k] (row-major 64x96 per layer),
// Wr[l][o][c] (row-major 32x32 per layer).
// ---------------------------------------------------------------------------
__global__ void wavenet_wconv(const float* __restrict__ dilW,
                              const float* __restrict__ resW,
                              unsigned short* __restrict__ Wd,
                              unsigned short* __restrict__ Wr)
{
    const int nd = NLAYERS * NGATED * 96;   // 122880
    const int nr = NLAYERS * NCH * NCH;     // 20480
    int idx = blockIdx.x * blockDim.x + threadIdx.x;
    if (idx < nd) {
        int col = idx % 96;
        int g   = (idx / 96) % NGATED;
        int l   = idx / (96 * NGATED);
        int c = col % 32, k = col / 32;
        float v = dilW[(((size_t)l * NGATED + g) * NCH + c) * 3 + k];
        Wd[idx] = cvt1_bf16(v);
    } else if (idx < nd + nr) {
        int j = idx - nd;                   // res_W flat [l][o][c][1] == [l][o*32+c]
        Wr[j] = cvt1_bf16(resW[j]);
    }
}

// ---------------------------------------------------------------------------
// Input 1x1 conv: act[b][t][c] = in_W[c]*x[b][t] + in_b[c], bf16 pairs.
// ---------------------------------------------------------------------------
__global__ void wavenet_inconv(const float* __restrict__ x,
                               const float* __restrict__ inW,
                               const float* __restrict__ inb,
                               unsigned int* __restrict__ act,  // dword = 2 bf16
                               int totalBT)
{
    int idx = blockIdx.x * blockDim.x + threadIdx.x;
    if (idx >= totalBT * 16) return;
    int p  = idx & 15;         // channel pair
    int bt = idx >> 4;
    float xv = x[bt];
    float a = inW[2 * p]     * xv + inb[2 * p];
    float b = inW[2 * p + 1] * xv + inb[2 * p + 1];
    act[(size_t)bt * 16 + p] = pack2_bf16(a, b);
}

__global__ void wavenet_init(float* __restrict__ out,
                             const float* __restrict__ pb, int n)
{
    int i = blockIdx.x * blockDim.x + threadIdx.x;
    if (i < n) out[i] = pb[0];
}

// ---------------------------------------------------------------------------
// One WaveNet layer. Each wave handles TWO 16-timestep tiles (32 timesteps):
//   24 WMMAs (dilated conv, 4 M-tiles x 3 taps x 2 N-tiles, A reused)
//   + gating + skip dot + 4 WMMAs (residual 1x1) + residual add + bf16 store.
// ---------------------------------------------------------------------------
__global__ __launch_bounds__(256) void wavenet_layer(
    const unsigned short* __restrict__ actIn,   // [B][T][32] bf16
    unsigned short* __restrict__ actOut,        // [B][T][32] bf16
    const unsigned short* __restrict__ Wd,      // [64][96] bf16 (this layer)
    const unsigned short* __restrict__ Wr,      // [32][32] bf16
    const float* __restrict__ dil_b,            // [64]
    const float* __restrict__ res_b,            // [32]
    const float* __restrict__ pw,               // [32] post_W row
    float* __restrict__ skipAcc,                // [B][T] (== d_out)
    int dil)
{
    const int lane = threadIdx.x & 31;
    const int wave = threadIdx.x >> 5;
    const int b    = blockIdx.y;
    const int t0   = (blockIdx.x * 8 + wave) * 32;   // 32 timesteps per wave
    const int n    = lane & 15;
    const int hi   = lane >> 4;                 // lane half selects K/M sub-block
    const int akoff = hi * 8;    // A operand: lanes 16-31 hold K=8-15 / 24-31

    const unsigned short* aIn  = actIn  + (size_t)b * T_LEN * NCH;
    unsigned short*       aOut = actOut + (size_t)b * T_LEN * NCH;

    // D/C layout: VGPR i, lanes 0-15 -> row i, lanes 16-31 -> row 8+i (time col n)
    v8f cacc[2][4];
    {
        const int rb = hi * 8;
#pragma unroll
        for (int i = 0; i < 8; ++i) {
            float b0 = dil_b[ 0 + rb + i];
            float b1 = dil_b[16 + rb + i];
            float b2 = dil_b[32 + rb + i];
            float b3 = dil_b[48 + rb + i];
            cacc[0][0][i] = b0; cacc[1][0][i] = b0;
            cacc[0][1][i] = b1; cacc[1][1][i] = b1;
            cacc[0][2][i] = b2; cacc[1][2][i] = b2;
            cacc[0][3][i] = b3; cacc[1][3][i] = b3;
        }
    }

    // ---- dilated conv: 3 taps (K-chunks) x 4 M-tiles x 2 N-tiles ----
#pragma unroll
    for (int k = 0; k < 3; ++k) {
        v16bf bmat[2];
#pragma unroll
        for (int u = 0; u < 2; ++u) {
            const int tt = t0 + 16 * u + n - (2 - k) * dil;   // causal tap offset
            v8u bu;
#pragma unroll
            for (int i = 0; i < 8; ++i) bu[i] = 0u;           // zero pad tt < 0
            if (tt >= 0) {
                // B layout: lanes 0-15 -> K(ch) 0-7 / 16-23, lanes 16-31 -> K 8-15 / 24-31
                const unsigned short* p = aIn + (size_t)tt * NCH + hi * 8;
                uint4 q0 = *(const uint4*)(p);
                uint4 q1 = *(const uint4*)(p + 16);
                bu[0] = q0.x; bu[1] = q0.y; bu[2] = q0.z; bu[3] = q0.w;
                bu[4] = q1.x; bu[5] = q1.y; bu[6] = q1.z; bu[7] = q1.w;
            }
            bmat[u] = __builtin_bit_cast(v16bf, bu);
        }
#pragma unroll
        for (int mt = 0; mt < 4; ++mt) {
            const unsigned short* wp = Wd + (size_t)(mt * 16 + n) * 96 + k * 32;
            uint4 w0 = *(const uint4*)(wp + akoff);
            uint4 w1 = *(const uint4*)(wp + 16 + akoff);
            v8u au;
            au[0] = w0.x; au[1] = w0.y; au[2] = w0.z; au[3] = w0.w;
            au[4] = w1.x; au[5] = w1.y; au[6] = w1.z; au[7] = w1.w;
            v16bf amat = __builtin_bit_cast(v16bf, au);
            cacc[0][mt] = __builtin_amdgcn_wmma_f32_16x16x32_bf16(
                false, amat, false, bmat[0], (short)0, cacc[0][mt], false, false);
            cacc[1][mt] = __builtin_amdgcn_wmma_f32_16x16x32_bf16(
                false, amat, false, bmat[1], (short)0, cacc[1][mt], false, false);
        }
    }

    // ---- residual A tiles (shared across both N-tiles) ----
    v16bf rA[2];
#pragma unroll
    for (int mt = 0; mt < 2; ++mt) {
        const unsigned short* wp = Wr + (size_t)(mt * 16 + n) * 32;
        uint4 w0 = *(const uint4*)(wp + akoff);
        uint4 w1 = *(const uint4*)(wp + 16 + akoff);
        v8u au;
        au[0] = w0.x; au[1] = w0.y; au[2] = w0.z; au[3] = w0.w;
        au[4] = w1.x; au[5] = w1.y; au[6] = w1.z; au[7] = w1.w;
        rA[mt] = __builtin_bit_cast(v16bf, au);
    }

#pragma unroll
    for (int u = 0; u < 2; ++u) {
        const int t = t0 + 16 * u + n;

        // ---- gated activation (elementwise in D layout, hw trans ops) ----
        v8f g0, g1;   // gated channels 0-15 / 16-31 (in D layout)
#pragma unroll
        for (int i = 0; i < 8; ++i) {
            g0[i] = hw_tanh(cacc[u][0][i]) * hw_sigmoid(cacc[u][2][i]);
            g1[i] = hw_tanh(cacc[u][1][i]) * hw_sigmoid(cacc[u][3][i]);
        }

        // ---- skip contribution: dot with post_W row, reduce lanes (L, L+16) --
        {
            const int off = hi * 8;
            float partial = 0.0f;
#pragma unroll
            for (int i = 0; i < 8; ++i)
                partial += pw[off + i] * g0[i] + pw[16 + off + i] * g1[i];
            partial += __shfl_xor(partial, 16, 32);
            if (hi == 0) {
                float* dst = skipAcc + (size_t)b * T_LEN + t;   // exclusive owner
                *dst = *dst + partial;
            }
        }

        // ---- repack gated D-layout -> res-conv B operand (bf16) ----
        unsigned int plo[4], phi[4];
#pragma unroll
        for (int j = 0; j < 4; ++j) {
            plo[j] = pack2_bf16(g0[2 * j], g0[2 * j + 1]);
            phi[j] = pack2_bf16(g1[2 * j], g1[2 * j + 1]);
        }
        v8u bru;
#pragma unroll
        for (int j = 0; j < 4; ++j) {
            unsigned int oplo = (unsigned int)__shfl_xor((int)plo[j], 16, 32);
            unsigned int ophi = (unsigned int)__shfl_xor((int)phi[j], 16, 32);
            bru[j]     = hi ? plo[j] : plo[j];  // K 0-7  / 8-15
            bru[4 + j] = hi ? phi[j] : phi[j];  // K 16-23 / 24-31
        }
        v16bf bres = __builtin_bit_cast(v16bf, bru);

        // ---- residual 1x1 conv + residual add, 2 M-tiles ----
#pragma unroll
        for (int mt = 0; mt < 2; ++mt) {
            const int cbase = mt * 16 + hi * 8;
            const uint4 r = *(const uint4*)(aIn + (size_t)t * NCH + cbase);
            v8f cc;
            cc[0] = bf16lo(r.x); cc[1] = bf16hi(r.x);
            cc[2] = bf16lo(r.y); cc[3] = bf16hi(r.y);
            cc[4] = bf16lo(r.z); cc[5] = bf16hi(r.z);
            cc[6] = bf16lo(r.w); cc[7] = bf16hi(r.w);
#pragma unroll
            for (int i = 0; i < 8; ++i) cc[i] += res_b[cbase + i];

            cc = __builtin_amdgcn_wmma_f32_16x16x32_bf16(
                false, rA[mt], false, bres, (short)0, cc, false, false);

            uint4 s;
            s.x = pack2_bf16(cc[0], cc[1]);
            s.y = pack2_bf16(cc[2], cc[3]);
            s.z = pack2_bf16(cc[4], cc[5]);
            s.w = pack2_bf16(cc[6], cc[7]);
            *(uint4*)(aOut + (size_t)t * NCH + cbase) = s;
        }
    }
}

// ---------------------------------------------------------------------------
extern "C" void kernel_launch(void* const* d_in, const int* in_sizes, int n_in,
                              void* d_out, int out_size, void* d_ws, size_t ws_size,
                              hipStream_t stream) {
    const float* x     = (const float*)d_in[0];
    const float* inW   = (const float*)d_in[1];
    const float* inb   = (const float*)d_in[2];
    const float* dilW  = (const float*)d_in[3];
    const float* dilb  = (const float*)d_in[4];
    const float* resW  = (const float*)d_in[5];
    const float* resb  = (const float*)d_in[6];
    const float* postW = (const float*)d_in[7];
    const float* postb = (const float*)d_in[8];
    float* out = (float*)d_out;

    const int B = in_sizes[0] / T_LEN;
    const size_t ACT = (size_t)B * T_LEN * NCH * 2;   // bf16 activations

    unsigned char* ws = (unsigned char*)d_ws;
    unsigned short* actA = (unsigned short*)(ws);
    unsigned short* actB = (unsigned short*)(ws + ACT);
    unsigned short* Wd   = (unsigned short*)(ws + 2 * ACT);
    unsigned short* Wr   = (unsigned short*)(ws + 2 * ACT +
                                             (size_t)NLAYERS * NGATED * 96 * 2);

    {   // weights -> bf16
        int tot = NLAYERS * NGATED * 96 + NLAYERS * NCH * NCH;
        wavenet_wconv<<<(tot + 255) / 256, 256, 0, stream>>>(dilW, resW, Wd, Wr);
    }
    {   // input 1x1 conv -> actA
        int totalBT = B * T_LEN;
        wavenet_inconv<<<(totalBT * 16 + 255) / 256, 256, 0, stream>>>(
            x, inW, inb, (unsigned int*)actA, totalBT);
    }
    // d_out starts at post_b, layer kernels accumulate skip contributions
    wavenet_init<<<(out_size + 255) / 256, 256, 0, stream>>>(out, postb, out_size);

    dim3 grid(T_LEN / 256, B);   // 8 waves/block * 32 t/wave = 256 t per block
    for (int l = 0; l < NLAYERS; ++l) {
        const int dil = 1 << (l % 10);
        const unsigned short* ain = (l & 1) ? actB : actA;
        unsigned short*       aout = (l & 1) ? actA : actB;
        wavenet_layer<<<grid, 256, 0, stream>>>(
            ain, aout,
            Wd + (size_t)l * NGATED * 96,
            Wr + (size_t)l * NCH * NCH,
            dilb + l * NGATED,
            resb + l * NCH,
            postW + l * NCH,
            out, dil);
    }
}
